// GCNGRU_nodes_22299470201221
// MI455X (gfx1250) — matrix operand order repaired and checked by reading, not compiled, not run on hardware
//
#include <hip/hip_runtime.h>
#include <hip/hip_bf16.h>
#include <math.h>

// ---------------------------------------------------------------------------
// GCN+GRU pipeline for MI455X (gfx1250, wave32).
// Dense GEMMs use v_wmma_f32_16x16x32_bf16 (one 16x16 C tile per wave).
// Sparse aggregation uses global_atomic_add_f32 scatter (L2-resident: all
// working sets fit in the 192MB global L2, so atomics never touch HBM).
// ---------------------------------------------------------------------------

typedef __bf16 v16bf __attribute__((ext_vector_type(16)));
typedef float  v8f   __attribute__((ext_vector_type(8)));

#define NGRU 64
#define NHID 128
#define NOUT 64

// ---------------- weight conversion: f32 -> bf16, optionally transposed ----
// Produces Bt[n][k] (n = output col of GEMM, k = reduction dim), contiguous
// in k so each WMMA lane's B fragment is a single 32-byte load.
__global__ void cvt_bf16_kernel(const float* __restrict__ src,
                                unsigned short* __restrict__ dst,
                                int Nc, int K, int trans) {
  int i = blockIdx.x * blockDim.x + threadIdx.x;
  if (i >= Nc * K) return;
  int n = i / K, k = i - n * K;
  float v = trans ? src[(size_t)k * Nc + n] : src[i];
  // round-to-nearest-even fp32 -> bf16
  unsigned int u = __float_as_uint(v);
  unsigned int r = (u + 0x7FFFu + ((u >> 16) & 1u)) >> 16;
  dst[i] = (unsigned short)r;
}

// ---------------- SpMM scatter: out[row,:] += val * src[col,:] -------------
template <int F>
__global__ void spmm_scatter_kernel(const int* __restrict__ adj_idx,
                                    const float* __restrict__ adj_val,
                                    const int* __restrict__ day_base, int dt,
                                    int E2, const float* __restrict__ src,
                                    float* __restrict__ out) {
  const int epb = 256 / F;
  const int e = blockIdx.x * epb + (int)(threadIdx.x / F);
  const int f = (int)(threadIdx.x & (F - 1));
  if (e >= E2) return;
  const int day = day_base[0] + dt;
  const int* idx = adj_idx + (size_t)day * 2 * E2;
  const int row = idx[e];
  const int col = idx[E2 + e];
  const float v = adj_val[(size_t)day * E2 + e];
  atomicAdd(&out[(size_t)row * F + f], v * src[(size_t)col * F + f]);
}

// ---------------- elementwise bias (+ optional relu) -----------------------
__global__ void bias_act_kernel(float* __restrict__ buf,
                                const float* __restrict__ bias,
                                int total, int Fm1, int relu) {
  int i = blockIdx.x * blockDim.x + threadIdx.x;
  if (i >= total) return;
  float v = buf[i] + bias[i & Fm1];
  buf[i] = relu ? fmaxf(v, 0.0f) : v;
}

// ---------------- WMMA GEMM: C = act(A[MxK] * Bt^T + bias) -----------------
// A: f32 row-major (lda = K), converted to bf16 in-register.
// Bt: bf16 [Nc][K] (i.e. B transposed) -> contiguous-k fragment loads.
// One wave computes one 16x16 tile of C. M, Nc multiples of 16.
__global__ void __launch_bounds__(128)
wmma_gemm_bf16_kernel(const float* __restrict__ A,
                      const unsigned short* __restrict__ Bt,
                      const float* __restrict__ bias,
                      float* __restrict__ C,
                      int M, int Nc, int K, int act) {
  const int lane = (int)(threadIdx.x & 31);
  const int wv   = (int)(threadIdx.x >> 5);
  const int tilesN = Nc >> 4;
  const int totalTiles = (M >> 4) * tilesN;
  const int tile = blockIdx.x * 4 + wv;
  if (tile >= totalTiles) return;           // wave-uniform: EXEC stays full

  const int tm = tile / tilesN;
  const int tn = tile - tm * tilesN;
  const int row0 = tm << 4, col0 = tn << 4;
  const int half = lane >> 4;               // K-half selector (contiguous-k order)
  const int l16  = lane & 15;

  v8f acc;
#pragma unroll
  for (int i = 0; i < 8; ++i) acc[i] = 0.0f;

  const float*  arow = A  + (size_t)(row0 + l16) * K;
  const __bf16* brow = (const __bf16*)(Bt + (size_t)(col0 + l16) * K) + 16 * half;

  for (int k0 = 0; k0 < K; k0 += 32) {
    // A fragment: contiguous-k order matching B:
    //   VGPR v: K = 2v..2v+1 (+16 for lanes 16-31)
    v16bf a;
#pragma unroll
    for (int v = 0; v < 8; ++v) {
      const int kl = 2 * v + 16 * half;
      const float2 f2 = *(const float2*)(arow + k0 + kl);
      a[2 * v]     = (__bf16)f2.x;
      a[2 * v + 1] = (__bf16)f2.y;
    }
    // B fragment: lane (&15) = column, K = j + 16*half -> contiguous 32B load
    const v16bf b = *(const v16bf*)(brow + k0);

    acc = __builtin_amdgcn_wmma_f32_16x16x32_bf16(
        /*neg_a=*/false, a, /*neg_b=*/false, b,
        /*c_mod=*/(short)0, acc, /*reuse_a=*/false, /*reuse_b=*/false);
  }

  const int col = col0 + l16;
  const float bcol = bias ? bias[col] : 0.0f;
#pragma unroll
  for (int r = 0; r < 8; ++r) {             // C layout: VGPR r -> row r+8*half
    const int row = row0 + r + 8 * half;
    float v = acc[r] + bcol;
    if (act) v = fmaxf(v, 0.0f);
    C[(size_t)row * Nc + col] = v;
  }
}

// ---------------- GRU gates ------------------------------------------------
__global__ void gru_gates_kernel(const float* __restrict__ gi,
                                 const float* __restrict__ gh,
                                 const float* __restrict__ hprev,
                                 float* __restrict__ hnext, int total) {
  int i = blockIdx.x * blockDim.x + threadIdx.x;
  if (i >= total) return;
  int n = i >> 6, c = i & 63;
  size_t b = (size_t)n * 192 + c;
  float r  = 1.0f / (1.0f + __expf(-(gi[b]       + gh[b])));
  float z  = 1.0f / (1.0f + __expf(-(gi[b + 64]  + gh[b + 64])));
  float ng = tanhf(gi[b + 128] + r * gh[b + 128]);
  hnext[i] = (1.0f - z) * ng + z * hprev[i];
}

// ---------------- batch-norm stats + apply ---------------------------------
__global__ void bn_stats_kernel(const float* __restrict__ h,
                                float* __restrict__ stats, int Nn) {
  const int col = (int)(threadIdx.x & 63);
  const int grp = (int)(threadIdx.x >> 6);
  float s = 0.0f, s2 = 0.0f;
  for (int n = blockIdx.x * 4 + grp; n < Nn; n += gridDim.x * 4) {
    float v = h[(size_t)n * 64 + col];
    s += v; s2 += v * v;
  }
  atomicAdd(&stats[col], s);
  atomicAdd(&stats[64 + col], s2);
}

__global__ void bn_apply_kernel(const float* __restrict__ h,
                                const float* __restrict__ stats,
                                const float* __restrict__ gamma,
                                const float* __restrict__ beta,
                                float* __restrict__ emb, int total, float invN) {
  int i = blockIdx.x * blockDim.x + threadIdx.x;
  if (i >= total) return;
  int c = i & 63;
  float mean = stats[c] * invN;
  float var  = stats[64 + c] * invN - mean * mean;
  emb[i] = (h[i] - mean) * rsqrtf(var + 1e-5f) * gamma[c] + beta[c];
}

// ---------------- attention aggregation ------------------------------------
__global__ void deg_kernel(const int* __restrict__ adj_idx,
                           const int* __restrict__ end_day, int E2,
                           float* __restrict__ deg) {
  int e = blockIdx.x * blockDim.x + threadIdx.x;
  if (e >= E2) return;
  const int day = end_day[0] + 1;
  const int* idx = adj_idx + (size_t)day * 2 * E2;
  int row = idx[e], col = idx[E2 + e];
  if (row != col) atomicAdd(&deg[row], 1.0f);
}

// one wave (32 lanes) per edge: dot over 128 concat-features, shfl-reduce,
// then scatter ev * emb[col] into neigh[row].
__global__ void attn_scatter_kernel(const int* __restrict__ adj_idx,
                                    const int* __restrict__ end_day, int E2,
                                    const float* __restrict__ emb,
                                    const float* __restrict__ attn_w,
                                    const float* __restrict__ attn_b,
                                    const float* __restrict__ deg,
                                    float* __restrict__ neigh) {
  const int lane = (int)(threadIdx.x & 31);
  const int e = blockIdx.x * 8 + (int)(threadIdx.x >> 5);
  if (e >= E2) return;
  const int day = end_day[0] + 1;
  const int* idx = adj_idx + (size_t)day * 2 * E2;
  const int row = idx[e], col = idx[E2 + e];
  const bool mask = (row != col);
  float p = emb[(size_t)row * 64 + lane]      * attn_w[lane]
          + emb[(size_t)row * 64 + 32 + lane] * attn_w[32 + lane]
          + emb[(size_t)col * 64 + lane]      * attn_w[64 + lane]
          + emb[(size_t)col * 64 + 32 + lane] * attn_w[96 + lane];
#pragma unroll
  for (int m = 16; m >= 1; m >>= 1) p += __shfl_xor(p, m, 32);
  const float wgt = 1.0f / (1.0f + __expf(-(p + attn_b[0])));
  const float d = deg[row];
  const float inv = (d != 0.0f) ? 1.0f / fmaxf(d, 1.0f) : 1.0f;
  const float ev = inv * wgt;
  if (mask) {
    atomicAdd(&neigh[(size_t)row * 64 + lane],      ev * emb[(size_t)col * 64 + lane]);
    atomicAdd(&neigh[(size_t)row * 64 + 32 + lane], ev * emb[(size_t)col * 64 + 32 + lane]);
  }
}

// ---------------- z = concat(emb, neigh) -----------------------------------
__global__ void concat_kernel(const float* __restrict__ emb,
                              const float* __restrict__ neigh,
                              float* __restrict__ z, int total) {
  int i = blockIdx.x * blockDim.x + threadIdx.x;
  if (i >= total) return;
  int n = i >> 7, c = i & 127;
  z[i] = (c < 64) ? emb[(size_t)n * 64 + c] : neigh[(size_t)n * 64 + (c - 64)];
}

// ---------------- final (64 -> 2) + log_softmax ----------------------------
__global__ void head_kernel(const float* __restrict__ h1,
                            const float* __restrict__ w2,
                            const float* __restrict__ b2v,
                            float* __restrict__ out, int Nn) {
  int n = blockIdx.x * blockDim.x + threadIdx.x;
  if (n >= Nn) return;
  const float* hr = h1 + (size_t)n * 64;
  float s0 = b2v[0], s1 = b2v[1];
#pragma unroll 8
  for (int k = 0; k < 64; ++k) {
    float h = hr[k];
    s0 += h * w2[k * 2 + 0];
    s1 += h * w2[k * 2 + 1];
  }
  float m = fmaxf(s0, s1);
  float lse = m + logf(__expf(s0 - m) + __expf(s1 - m));
  out[(size_t)n * 2 + 0] = s0 - lse;
  out[(size_t)n * 2 + 1] = s1 - lse;
}

// ===========================================================================
extern "C" void kernel_launch(void* const* d_in, const int* in_sizes, int n_in,
                              void* d_out, int out_size, void* d_ws, size_t ws_size,
                              hipStream_t stream) {
  const int*   adj_idx  = (const int*)  d_in[0];
  const float* adj_val  = (const float*)d_in[1];
  const int*   startd   = (const int*)  d_in[2];
  const int*   endd     = (const int*)  d_in[3];
  const float* W1       = (const float*)d_in[4];
  const float* b1       = (const float*)d_in[5];
  const float* W2       = (const float*)d_in[6];
  const float* b2       = (const float*)d_in[7];
  const float* w_ih     = (const float*)d_in[8];
  const float* w_hh     = (const float*)d_in[9];
  const float* b_ih     = (const float*)d_in[10];
  const float* b_hh     = (const float*)d_in[11];
  const float* bn_gamma = (const float*)d_in[12];
  const float* bn_beta  = (const float*)d_in[13];
  const float* attn_w   = (const float*)d_in[14];
  const float* attn_b   = (const float*)d_in[15];
  const float* np_w1    = (const float*)d_in[16];
  const float* np_b1    = (const float*)d_in[17];
  const float* np_w2    = (const float*)d_in[18];
  const float* np_b2    = (const float*)d_in[19];

  const int E2 = in_sizes[1] / 8;          // adj_val is (S=8, E)
  const int Nn = in_sizes[4] / NHID;       // W1 is (N, 128); N = 50000 (16*3125)
  const int T  = 7;

  // ---- workspace layout (floats) ----
  float* ws = (float*)d_ws;
  size_t off = 0;
  float* X     = ws + off; off += (size_t)Nn * 128;   // spmm1 accum / relu x / z
  float* XS    = ws + off; off += (size_t)Nn * 64;    // spmm2 accum + b2
  float* H     = ws + off; off += (size_t)Nn * 64;    // GRU state ping
  float* H2    = ws + off; off += (size_t)Nn * 64;    // GRU state pong
  float* GI    = ws + off; off += (size_t)Nn * 192;   // gi (also XW / emb / neigh reuse)
  float* GH    = ws + off; off += (size_t)Nn * 192;   // gh (also h1 reuse)
  float* DEG   = ws + off; off += (size_t)Nn;
  float* STATS = ws + off; off += 128;
  unsigned short* WB = (unsigned short*)(ws + off);   // bf16 weights
  unsigned short* wihT  = WB;                         // 192*64
  unsigned short* whhT  = WB + 192 * 64;              // 192*64
  unsigned short* w2t   = whhT + 192 * 64;            // 64*128 ([n][k])
  unsigned short* npw1t = w2t + 64 * 128;             // 64*128 ([n][k])
  const size_t needed = off * sizeof(float) + (size_t)40960 * sizeof(unsigned short);
  if (ws_size < needed) return;

  float* XW    = GI;            // x@W2 result, dead before gi GEMM writes GI
  float* EMB   = GI;            // normalized final_emb (post-steps)
  float* NEIGH = GI + (size_t)Nn * 64;
  float* H1    = GH;            // relu MLP hidden (post-steps)

  // ---- convert weights to bf16 B^T layout ----
  cvt_bf16_kernel<<<(192 * 64 + 255) / 256, 256, 0, stream>>>(w_ih, wihT, 192, 64, 0);
  cvt_bf16_kernel<<<(192 * 64 + 255) / 256, 256, 0, stream>>>(w_hh, whhT, 192, 64, 0);
  cvt_bf16_kernel<<<(64 * 128 + 255) / 256, 256, 0, stream>>>(W2,    w2t,   64, 128, 1);
  cvt_bf16_kernel<<<(64 * 128 + 255) / 256, 256, 0, stream>>>(np_w1, npw1t, 64, 128, 1);

  hipMemsetAsync(H, 0, (size_t)Nn * 64 * sizeof(float), stream);   // h0 = 0

  float* hp = H;
  float* hn = H2;
  const int elems64  = Nn * 64;
  const int elems128 = Nn * 128;
  const int gemmTiles64  = (Nn >> 4) * (64 >> 4);    // Nc=64
  const int gemmTiles192 = (Nn >> 4) * (192 >> 4);   // Nc=192

  for (int t = 0; t < T; ++t) {
    // x = relu(spmm(A_t, W1) + b1)
    hipMemsetAsync(X, 0, (size_t)elems128 * sizeof(float), stream);
    spmm_scatter_kernel<128><<<(E2 + 1) / 2, 256, 0, stream>>>(
        adj_idx, adj_val, startd, t, E2, W1, X);
    bias_act_kernel<<<(elems128 + 255) / 256, 256, 0, stream>>>(X, b1, elems128, 127, 1);

    // xw = x @ W2   (WMMA, Nc=64, K=128)
    wmma_gemm_bf16_kernel<<<(gemmTiles64 + 3) / 4, 128, 0, stream>>>(
        X, w2t, nullptr, XW, Nn, 64, 128, 0);

    // x2 = spmm(A_t, xw) + b2
    hipMemsetAsync(XS, 0, (size_t)elems64 * sizeof(float), stream);
    spmm_scatter_kernel<64><<<(E2 + 3) / 4, 256, 0, stream>>>(
        adj_idx, adj_val, startd, t, E2, XW, XS);
    bias_act_kernel<<<(elems64 + 255) / 256, 256, 0, stream>>>(XS, b2, elems64, 63, 0);

    // gi = x2 @ w_ih^T + b_ih ; gh = h @ w_hh^T + b_hh   (WMMA, Nc=192, K=64)
    wmma_gemm_bf16_kernel<<<(gemmTiles192 + 3) / 4, 128, 0, stream>>>(
        XS, wihT, b_ih, GI, Nn, 192, 64, 0);
    wmma_gemm_bf16_kernel<<<(gemmTiles192 + 3) / 4, 128, 0, stream>>>(
        hp, whhT, b_hh, GH, Nn, 192, 64, 0);

    gru_gates_kernel<<<(elems64 + 255) / 256, 256, 0, stream>>>(GI, GH, hp, hn, elems64);
    float* tmp = hp; hp = hn; hn = tmp;
  }
  // final state is in hp

  // ---- batch norm over nodes ----
  hipMemsetAsync(STATS, 0, 128 * sizeof(float), stream);
  bn_stats_kernel<<<512, 256, 0, stream>>>(hp, STATS, Nn);
  bn_apply_kernel<<<(elems64 + 255) / 256, 256, 0, stream>>>(
      hp, STATS, bn_gamma, bn_beta, EMB, elems64, 1.0f / (float)Nn);

  // ---- attention neighbor aggregation on day end+1 ----
  hipMemsetAsync(DEG, 0, (size_t)Nn * sizeof(float), stream);
  hipMemsetAsync(NEIGH, 0, (size_t)elems64 * sizeof(float), stream);
  deg_kernel<<<(E2 + 255) / 256, 256, 0, stream>>>(adj_idx, endd, E2, DEG);
  attn_scatter_kernel<<<(E2 + 7) / 8, 256, 0, stream>>>(
      adj_idx, endd, E2, EMB, attn_w, attn_b, DEG, NEIGH);

  // ---- head: z = concat -> relu(z@np_w1+b) (WMMA) -> (64->2) log_softmax ----
  concat_kernel<<<(elems128 + 255) / 256, 256, 0, stream>>>(EMB, NEIGH, X, elems128);
  wmma_gemm_bf16_kernel<<<(gemmTiles64 + 3) / 4, 128, 0, stream>>>(
      X, npw1t, np_b1, H1, Nn, 64, 128, 1);
  head_kernel<<<(Nn + 255) / 256, 256, 0, stream>>>(H1, np_w2, np_b2, (float*)d_out, Nn);
}
